// RotQuantWrapper_59296318488893
// MI455X (gfx1250) — matrix operand order repaired and checked by reading, not compiled, not run on hardware
//
#include <hip/hip_runtime.h>
#include <hip/hip_bf16.h>
#include <stdint.h>

typedef unsigned short u16;
typedef unsigned int   u32;
typedef __attribute__((ext_vector_type(16))) __bf16 v16bf;
typedef __attribute__((ext_vector_type(8)))  float  v8f;

#define OUT_FEAT 11008
#define IN_FEAT  4096
#define HAD_DIM  128

#if __has_builtin(__builtin_amdgcn_global_load_async_to_lds_b128) && \
    __has_builtin(__builtin_amdgcn_s_wait_asynccnt)
#define HAVE_ASYNC_LDS 1
#else
#define HAVE_ASYNC_LDS 0
#endif

// ---------- helpers ----------

__device__ __forceinline__ u16 f2bf(float f) {
  u32 u = __float_as_uint(f);
  u32 r = u + 0x7FFFu + ((u >> 16) & 1u);   // round-to-nearest-even
  return (u16)(r >> 16);
}

union FragBF { v16bf v; u32 u[8]; };

// A-fragment (16x32 bf16), per ISA layout:
// lane l: M = l%16, half h = l/16.
// VGPR j(0..3) holds K = 2j + 8h (pair), VGPR j(4..7) holds K = 2(j-4)+16+8h (pair).
__device__ __forceinline__ v16bf load_a_frag(const u16* As, int ldA, int m0, int kk, int lane) {
  const int m = m0 + (lane & 15);
  const int h = lane >> 4;
  const u32* row = (const u32*)(As + (size_t)m * ldA + kk);  // kk even
  FragBF f;
#pragma unroll
  for (int j = 0; j < 4; ++j) f.u[j]     = row[j + 4 * h];
#pragma unroll
  for (int j = 0; j < 4; ++j) f.u[4 + j] = row[8 + j + 4 * h];
  return f.v;
}

// B-fragment (32x16 bf16): lane l holds column N = l%16; VGPR j(0..3) K = 2j + 8h, j(4..7) K = 2(j-4)+16+8h (pairs).
__device__ __forceinline__ v16bf load_b_frag(const u16* Bs, int ldB, int kk, int n0, int lane) {
  const int n = n0 + (lane & 15);
  const int h = lane >> 4;
  FragBF f;
#pragma unroll
  for (int j = 0; j < 8; ++j) {
    const int k0 = kk + (j < 4 ? 2 * j : 16 + 2 * (j - 4)) + 8 * h;
    f.u[j] = (u32)Bs[(size_t)k0 * ldB + n] | ((u32)Bs[(size_t)(k0 + 1) * ldB + n] << 16);
  }
  return f.v;
}

__device__ __forceinline__ v8f wmma_bf16(v16bf a, v16bf b, v8f c) {
  return __builtin_amdgcn_wmma_f32_16x16x32_bf16(false, a, false, b, (short)0, c, false, false);
}

#if HAVE_ASYNC_LDS
// Async 16-byte global -> LDS copy, tracked by ASYNCcnt.
// Builtin signature (from hipcc diagnostic): pointee is vector_size(16) int.
typedef int v4i_vs __attribute__((vector_size(16)));
typedef __attribute__((address_space(1))) v4i_vs* g_b128_ptr;
typedef __attribute__((address_space(3))) v4i_vs* l_b128_ptr;

__device__ __forceinline__ void async_copy_b128(const u16* g, u16* l) {
  __builtin_amdgcn_global_load_async_to_lds_b128((g_b128_ptr)(g), (l_b128_ptr)(l), 0, 0);
}
#endif

// ---------- conversion kernels ----------

// R_left fp32 -> bf16, 4 elements/thread (4096*4096 / (256*4) = 16384 blocks)
__global__ __launch_bounds__(256) void cvt_rl(const float* __restrict__ src, u16* __restrict__ dst) {
  const int i = blockIdx.x * 256 + threadIdx.x;
  float4 v = ((const float4*)src)[i];
  u32 lo = (u32)f2bf(v.x) | ((u32)f2bf(v.y) << 16);
  u32 hi = (u32)f2bf(v.z) | ((u32)f2bf(v.w) << 16);
  ((u32*)dst)[2 * i]     = lo;
  ((u32*)dst)[2 * i + 1] = hi;
}

// RrT[r*128 + k] = bf16(R_right[k*128 + r])   (64 blocks x 256 threads)
__global__ __launch_bounds__(256) void build_rrt(const float* __restrict__ rr, u16* __restrict__ rrt) {
  const int idx = blockIdx.x * 256 + threadIdx.x;
  const int r = idx >> 7, k = idx & 127;
  rrt[idx] = f2bf(rr[k * HAD_DIM + r]);
}

// ---------- GEMM 1: W2 = blockdiag(R_right^T) @ inp  (bf16 out) ----------
// grid: (IN_FEAT/128 = 32, OUT_FEAT/128 = 86), block 256 threads (8 waves)
__global__ __launch_bounds__(256) void gemm_rot(const u16* __restrict__ RrT,
                                                const float* __restrict__ inp,
                                                u16* __restrict__ W2) {
  __shared__ __align__(16) u16 Ar[HAD_DIM][HAD_DIM];  // 32 KB: R_right^T (M=r, K=k)
  __shared__ __align__(16) u16 Bi[HAD_DIM][HAD_DIM];  // 32 KB: inp block, bf16 (K=k, N=i)

  const int b  = blockIdx.y;          // which 128-row block of out_feat
  const int bn = blockIdx.x * 128;    // column (in_feat) tile
  const int t  = threadIdx.x;

  // stage A: 16384 u16 = 2048 uint4; 8 per thread
  {
    const uint4* src = (const uint4*)RrT;
    uint4* dst = (uint4*)&Ar[0][0];
#pragma unroll
    for (int j = 0; j < 8; ++j) dst[t + 256 * j] = src[t + 256 * j];
  }
  // stage B with fp32->bf16 conversion: 2 threads per row, 64 cols each
  {
    const int k  = t >> 1;
    const int c0 = (t & 1) * 64;
    const float* src = inp + (size_t)(b * 128 + k) * IN_FEAT + bn + c0;
    u32* dstrow = (u32*)&Bi[k][c0];
#pragma unroll
    for (int j = 0; j < 16; ++j) {
      float4 v = ((const float4*)src)[j];
      dstrow[2 * j]     = (u32)f2bf(v.x) | ((u32)f2bf(v.y) << 16);
      dstrow[2 * j + 1] = (u32)f2bf(v.z) | ((u32)f2bf(v.w) << 16);
    }
  }
  __syncthreads();

  const int lane = t & 31;
  const int wid  = t >> 5;
  const int wm   = (wid & 3) * 32;   // 4 waves along M: 2 tiles each
  const int wn   = (wid >> 2) * 64;  // 2 waves along N: 4 tiles each

  v8f acc[2][4];
#pragma unroll
  for (int mt = 0; mt < 2; ++mt)
#pragma unroll
    for (int nt = 0; nt < 4; ++nt) acc[mt][nt] = (v8f)(0.0f);

#pragma unroll
  for (int kk = 0; kk < HAD_DIM; kk += 32) {
    v16bf af[2], bfr[4];
#pragma unroll
    for (int mt = 0; mt < 2; ++mt) af[mt]  = load_a_frag(&Ar[0][0], HAD_DIM, wm + mt * 16, kk, lane);
#pragma unroll
    for (int nt = 0; nt < 4; ++nt) bfr[nt] = load_b_frag(&Bi[0][0], HAD_DIM, kk, wn + nt * 16, lane);
#pragma unroll
    for (int mt = 0; mt < 2; ++mt)
#pragma unroll
      for (int nt = 0; nt < 4; ++nt) acc[mt][nt] = wmma_bf16(af[mt], bfr[nt], acc[mt][nt]);
  }

  // write bf16 result: C layout VGPR r -> M = r + 8h, N = lane%16
  const int h = lane >> 4, nl = lane & 15;
#pragma unroll
  for (int mt = 0; mt < 2; ++mt) {
#pragma unroll
    for (int nt = 0; nt < 4; ++nt) {
      const int col = bn + wn + nt * 16 + nl;
#pragma unroll
      for (int r = 0; r < 8; ++r) {
        const int grow = b * 128 + wm + mt * 16 + r + 8 * h;
        W2[(size_t)grow * IN_FEAT + col] = f2bf(acc[mt][nt][r]);
      }
    }
  }
}

// ---------- GEMM 2: Y = W2 @ R_left (fp32 out), double-buffered async staging ----------
// grid: (4096/128 = 32, 11008/128 = 86), block 256 threads (8 waves)
__global__ __launch_bounds__(256) void gemm_main(const u16* __restrict__ Ag,
                                                 const u16* __restrict__ Bg,
                                                 float* __restrict__ out) {
  __shared__ __align__(16) u16 As[2][128][32];   // 2 x 8 KB
  __shared__ __align__(16) u16 Bs[2][32][128];   // 2 x 8 KB

  const int bn = blockIdx.x * 128;
  const int bm = blockIdx.y * 128;
  const int t  = threadIdx.x;
  const int lane = t & 31;
  const int wid  = t >> 5;
  const int wm   = (wid & 3) * 32;
  const int wn   = (wid >> 2) * 64;

  v8f acc[2][4];
#pragma unroll
  for (int mt = 0; mt < 2; ++mt)
#pragma unroll
    for (int nt = 0; nt < 4; ++nt) acc[mt][nt] = (v8f)(0.0f);

  const int arow = t >> 1, ahalf = (t & 1) * 16;  // 2 threads/row, 16 elems each
  const int brow = t >> 3, bseg  = (t & 7) * 16;  // 8 threads/row, 16 elems each

  const u16* agbase = Ag + (size_t)(bm + arow) * IN_FEAT + ahalf;
  const u16* bgbase = Bg + (size_t)brow * IN_FEAT + bn + bseg;

#if HAVE_ASYNC_LDS
  // ---- async double-buffered pipeline ----
  async_copy_b128(agbase + 0, &As[0][arow][ahalf]);
  async_copy_b128(agbase + 8, &As[0][arow][ahalf + 8]);
  async_copy_b128(bgbase + 0, &Bs[0][brow][bseg]);
  async_copy_b128(bgbase + 8, &Bs[0][brow][bseg + 8]);
  __builtin_amdgcn_s_wait_asynccnt(0);
  __syncthreads();

  int cur = 0;
  for (int kk = 0; kk < IN_FEAT; kk += 32) {
    const int nxt = cur ^ 1;
    if (kk + 32 < IN_FEAT) {
      const u16* an  = agbase + (kk + 32);
      const u16* bnp = bgbase + (size_t)(kk + 32) * IN_FEAT;
      async_copy_b128(an + 0, &As[nxt][arow][ahalf]);
      async_copy_b128(an + 8, &As[nxt][arow][ahalf + 8]);
      async_copy_b128(bnp + 0, &Bs[nxt][brow][bseg]);
      async_copy_b128(bnp + 8, &Bs[nxt][brow][bseg + 8]);
    }

    v16bf af[2], bfr[4];
#pragma unroll
    for (int mt = 0; mt < 2; ++mt) af[mt]  = load_a_frag(&As[cur][0][0], 32, wm + mt * 16, 0, lane);
#pragma unroll
    for (int nt = 0; nt < 4; ++nt) bfr[nt] = load_b_frag(&Bs[cur][0][0], 128, 0, wn + nt * 16, lane);
#pragma unroll
    for (int mt = 0; mt < 2; ++mt)
#pragma unroll
      for (int nt = 0; nt < 4; ++nt) acc[mt][nt] = wmma_bf16(af[mt], bfr[nt], acc[mt][nt]);

    if (kk + 32 < IN_FEAT) __builtin_amdgcn_s_wait_asynccnt(0);
    __syncthreads();
    cur = nxt;
  }
#else
  // ---- fallback: synchronous staging (single buffer) ----
  for (int kk = 0; kk < IN_FEAT; kk += 32) {
    const uint4* asrc = (const uint4*)(agbase + kk);
    *(uint4*)&As[0][arow][ahalf]     = asrc[0];
    *(uint4*)&As[0][arow][ahalf + 8] = asrc[1];
    const uint4* bsrc = (const uint4*)(bgbase + (size_t)kk * IN_FEAT);
    *(uint4*)&Bs[0][brow][bseg]     = bsrc[0];
    *(uint4*)&Bs[0][brow][bseg + 8] = bsrc[1];
    __syncthreads();

    v16bf af[2], bfr[4];
#pragma unroll
    for (int mt = 0; mt < 2; ++mt) af[mt]  = load_a_frag(&As[0][0][0], 32, wm + mt * 16, 0, lane);
#pragma unroll
    for (int nt = 0; nt < 4; ++nt) bfr[nt] = load_b_frag(&Bs[0][0][0], 128, 0, wn + nt * 16, lane);
#pragma unroll
    for (int mt = 0; mt < 2; ++mt)
#pragma unroll
      for (int nt = 0; nt < 4; ++nt) acc[mt][nt] = wmma_bf16(af[mt], bfr[nt], acc[mt][nt]);
    __syncthreads();
  }
#endif

  const int h = lane >> 4, nl = lane & 15;
#pragma unroll
  for (int mt = 0; mt < 2; ++mt) {
#pragma unroll
    for (int nt = 0; nt < 4; ++nt) {
      const int col = bn + wn + nt * 16 + nl;
#pragma unroll
      for (int r = 0; r < 8; ++r) {
        const int grow = bm + wm + mt * 16 + r + 8 * h;
        out[(size_t)grow * IN_FEAT + col] = acc[mt][nt][r];
      }
    }
  }
}

// ---------- fake quant ----------

__global__ void absmax_init(u32* amax) { *amax = 0u; }

__global__ __launch_bounds__(256) void absmax_kernel(const float4* __restrict__ x,
                                                     u32* __restrict__ amax, int n4) {
  int i = blockIdx.x * blockDim.x + threadIdx.x;
  const int stride = gridDim.x * blockDim.x;
  float m = 0.0f;
  for (; i < n4; i += stride) {
    if (i + stride < n4) __builtin_prefetch(&x[i + stride], 0, 0);  // global_prefetch_b8
    float4 v = x[i];
    m = fmaxf(m, fmaxf(fmaxf(fabsf(v.x), fabsf(v.y)), fmaxf(fabsf(v.z), fabsf(v.w))));
  }
#pragma unroll
  for (int off = 16; off > 0; off >>= 1) m = fmaxf(m, __shfl_down(m, off, 32));
  if ((threadIdx.x & 31) == 0) atomicMax(amax, __float_as_uint(m));  // all values >= 0
}

__global__ __launch_bounds__(256) void quant_kernel(float4* __restrict__ x,
                                                    const u32* __restrict__ amax, int n4) {
  const float amaxv = __uint_as_float(*amax);
  const float scale = fmaxf(amaxv * (1.0f / 127.0f), 1.17549435e-38f);  // finfo(f32).tiny
  const float inv = 1.0f / scale;
  int i = blockIdx.x * blockDim.x + threadIdx.x;
  const int stride = gridDim.x * blockDim.x;
  for (; i < n4; i += stride) {
    float4 v = x[i];
    v.x = fminf(fmaxf(rintf(v.x * inv), -127.0f), 127.0f) * scale;
    v.y = fminf(fmaxf(rintf(v.y * inv), -127.0f), 127.0f) * scale;
    v.z = fminf(fmaxf(rintf(v.z * inv), -127.0f), 127.0f) * scale;
    v.w = fminf(fmaxf(rintf(v.w * inv), -127.0f), 127.0f) * scale;
    x[i] = v;
  }
}

// ---------- launch ----------

extern "C" void kernel_launch(void* const* d_in, const int* in_sizes, int n_in,
                              void* d_out, int out_size, void* d_ws, size_t ws_size,
                              hipStream_t stream) {
  const float* inp = (const float*)d_in[0];   // (11008, 4096) fp32
  const float* RL  = (const float*)d_in[1];   // (4096, 4096) fp32
  const float* RR  = (const float*)d_in[2];   // (128, 128)   fp32
  float* out = (float*)d_out;                 // (11008, 4096) fp32

  char* ws = (char*)d_ws;
  const size_t W2_BYTES  = (size_t)OUT_FEAT * IN_FEAT * 2;  // 90,177,536
  const size_t RLB_BYTES = (size_t)IN_FEAT * IN_FEAT * 2;   // 33,554,432
  const size_t RRT_BYTES = (size_t)HAD_DIM * HAD_DIM * 2;   // 32,768
  u16* W2   = (u16*)ws;
  u16* RLb  = (u16*)(ws + W2_BYTES);
  u16* RrT  = (u16*)(ws + W2_BYTES + RLB_BYTES);
  u32* amax = (u32*)(ws + W2_BYTES + RLB_BYTES + RRT_BYTES);

  absmax_init<<<1, 1, 0, stream>>>(amax);
  cvt_rl<<<(IN_FEAT * IN_FEAT) / (256 * 4), 256, 0, stream>>>(RL, RLb);
  build_rrt<<<(HAD_DIM * HAD_DIM) / 256, 256, 0, stream>>>(RR, RrT);

  dim3 grot(IN_FEAT / 128, OUT_FEAT / 128);   // (32, 86)
  gemm_rot<<<grot, 256, 0, stream>>>(RrT, inp, W2);

  dim3 gmain(IN_FEAT / 128, OUT_FEAT / 128);  // (32, 86)
  gemm_main<<<gmain, 256, 0, stream>>>(W2, RLb, out);

  const int n4 = (OUT_FEAT * IN_FEAT) / 4;
  absmax_kernel<<<2048, 256, 0, stream>>>((const float4*)out, amax, n4);
  quant_kernel<<<2048, 256, 0, stream>>>((float4*)out, amax, n4);
}
